// UFO_Attention_2688649528044
// MI455X (gfx1250) — matrix operand order repaired and checked by reading, not compiled, not run on hardware
//
#include <hip/hip_runtime.h>
#include <hip/hip_bf16.h>

typedef __attribute__((ext_vector_type(16))) _Float16 v16h;
typedef __attribute__((ext_vector_type(8)))  _Float16 v8h;
typedef __attribute__((ext_vector_type(4)))  _Float16 v4h;
typedef __attribute__((ext_vector_type(8)))  float    v8f;
typedef __attribute__((ext_vector_type(4)))  float    v4f;

// ---------------------------------------------------------------------------
// Fragment gather helpers (CDNA5 wave32 WMMA f16 16x16x32 VGPR layouts).
// A-matrix 16x32: lane<16 holds M=lane, K{0..7}U{16..23}; lane>=16 holds
//                 M=lane-16, K{8..15}U{24..31}  (interleaved halves).
// B-matrix 32x16: lane<16 holds N=lane, K{0..7}U{16..23}; lane>=16 holds
//                 N=lane-16, K{8..15}U{24..31}  (interleaved halves).
// ---------------------------------------------------------------------------
__device__ __forceinline__ v16h ldfragA(const _Float16* p) {
    v8h lo = *(const v8h*)p;          // K base .. base+7
    v8h hi = *(const v8h*)(p + 16);   // K base+16 .. base+23
    return __builtin_shufflevector(lo, hi, 0,1,2,3,4,5,6,7,8,9,10,11,12,13,14,15);
}
__device__ __forceinline__ v16h ldfragB(const _Float16* p) {
    v8h lo = *(const v8h*)p;          // K base .. base+7
    v8h hi = *(const v8h*)(p + 16);   // K base+16 .. base+23
    return __builtin_shufflevector(lo, hi, 0,1,2,3,4,5,6,7,8,9,10,11,12,13,14,15);
}
__device__ __forceinline__ v8f wmma_f16(v16h a, v16h b, v8f c) {
    return __builtin_amdgcn_wmma_f32_16x16x32_f16(false, a, false, b, (short)0, c,
                                                  false, false);
}

// ---------------------------------------------------------------------------
// CDNA5 async memory->LDS (ASYNCcnt-tracked; no VGPR round trip).
// Generic LDS pointer low 32 bits == wave-relative LDS byte address.
// ---------------------------------------------------------------------------
__device__ __forceinline__ uint32_t lds_off(const void* p) {
    return (uint32_t)(uintptr_t)p;
}
__device__ __forceinline__ void async_load_b128(uint32_t lds, const void* g) {
    asm volatile("global_load_async_to_lds_b128 %0, %1, off"
                 :: "v"(lds), "v"((uint64_t)(uintptr_t)g) : "memory");
}
__device__ __forceinline__ void wait_async() {
    asm volatile("s_wait_asynccnt 0x0" ::: "memory");
}

// ---------------------------------------------------------------------------
// f32 -> f16 conversion (vectorized by 4; all sizes are multiples of 4)
// ---------------------------------------------------------------------------
__global__ __launch_bounds__(256) void cvt_f32_f16(const float* __restrict__ in,
                                                   _Float16* __restrict__ out, int n) {
    int i = (blockIdx.x * 256 + threadIdx.x) * 4;
    if (i < n) {
        v4f v = *(const v4f*)(in + i);
        v4h o;
        o[0] = (_Float16)v[0]; o[1] = (_Float16)v[1];
        o[2] = (_Float16)v[2]; o[3] = (_Float16)v[3];
        *(v4h*)(out + i) = o;
    }
}

// ---------------------------------------------------------------------------
// Tiled WMMA GEMM: C[M,N] = A[M,K] @ B[K,N], f16 in, f32 accumulate.
// Block = 256 threads (8 waves), tile 128x128, K-step 32, double-buffered LDS.
// A tile staged via global_load_async_to_lds_b128 (ASYNCcnt); B tile staged
// through VGPRs with a transpose scatter into [n][k] LDS layout.
// Wave (wm,wn) owns 32x64: 2 M-frags x 4 N-frags.
// ---------------------------------------------------------------------------
template <bool OUT_F32>
__global__ __launch_bounds__(256) void gemm_f16_wmma(
    const _Float16* __restrict__ A, const _Float16* __restrict__ B,
    _Float16* __restrict__ Ch, float* __restrict__ Cf,
    const float* __restrict__ bias, int M, int N, int K) {
    __shared__ alignas(16) _Float16 As[2][128 * 32];   // [m][k]
    __shared__ alignas(16) _Float16 Bs[2][128 * 32];   // transposed: [n][k]

    const int tid  = threadIdx.x;
    const int lane = tid & 31;
    const int wave = tid >> 5;
    const int wm   = wave >> 1;            // 0..3
    const int wn   = wave & 1;             // 0..1
    const int m0   = blockIdx.y * 128;
    const int n0   = blockIdx.x * 128;
    const int mrow = lane & 15;
    const int kbA  = (lane < 16) ? 0 : 8;
    const int kbB  = (lane < 16) ? 0 : 8;

    // Per-thread staging geometry (2 x b128 per thread per tile)
    int rA[2], cgA[2], krB[2], ngB[2];
    #pragma unroll
    for (int i = 0; i < 2; ++i) {
        int idx = tid * 2 + i;
        rA[i] = idx >> 2;  cgA[i] = idx & 3;    // A: 128 rows x 4 groups
        krB[i] = idx >> 4; ngB[i] = idx & 15;   // B: 32 rows x 16 groups
    }

    v8f acc[2][4] = {};

    auto stage_a = [&](int kt, int b) {
        #pragma unroll
        for (int i = 0; i < 2; ++i)
            async_load_b128(lds_off(&As[b][rA[i] * 32 + cgA[i] * 8]),
                            A + (size_t)(m0 + rA[i]) * K + kt + cgA[i] * 8);
    };

    // Prologue: stage tile kt=0 into buffer 0
    stage_a(0, 0);
    v8h pb[2];
    #pragma unroll
    for (int i = 0; i < 2; ++i)
        pb[i] = *(const v8h*)(B + (size_t)krB[i] * N + n0 + ngB[i] * 8);
    #pragma unroll
    for (int i = 0; i < 2; ++i)
        #pragma unroll
        for (int j = 0; j < 8; ++j)
            Bs[0][(ngB[i] * 8 + j) * 32 + krB[i]] = pb[i][j];
    wait_async();
    __syncthreads();

    int buf = 0;
    for (int kt = 0; kt < K; kt += 32) {
        const bool nxt = (kt + 32) < K;
        if (nxt) {
            // Issue next tile's loads; no waits until after compute.
            stage_a(kt + 32, buf ^ 1);
            #pragma unroll
            for (int i = 0; i < 2; ++i)
                pb[i] = *(const v8h*)(B + (size_t)(kt + 32 + krB[i]) * N +
                                      n0 + ngB[i] * 8);
        }
        if (kt + 64 < K) {   // prime L2 two tiles ahead
            __builtin_prefetch(A + (size_t)(m0 + rA[0]) * K + kt + 64, 0, 0);
            __builtin_prefetch(B + (size_t)(kt + 64 + krB[0]) * N + n0 + ngB[0] * 8, 0, 0);
        }

        // Compute on current buffer (overlaps the in-flight loads above)
        v16h af[2], bf[4];
        #pragma unroll
        for (int mi = 0; mi < 2; ++mi)
            af[mi] = ldfragA(&As[buf][(wm * 32 + mi * 16 + mrow) * 32 + kbA]);
        #pragma unroll
        for (int ni = 0; ni < 4; ++ni)
            bf[ni] = ldfragB(&Bs[buf][(wn * 64 + ni * 16 + mrow) * 32 + kbB]);
        #pragma unroll
        for (int mi = 0; mi < 2; ++mi)
            #pragma unroll
            for (int ni = 0; ni < 4; ++ni)
                acc[mi][ni] = wmma_f16(af[mi], bf[ni], acc[mi][ni]);

        if (nxt) {  // commit B tile (waits loadcnt only here, after the math)
            #pragma unroll
            for (int i = 0; i < 2; ++i)
                #pragma unroll
                for (int j = 0; j < 8; ++j)
                    Bs[buf ^ 1][(ngB[i] * 8 + j) * 32 + krB[i]] = pb[i][j];
        }
        wait_async();
        __syncthreads();
        buf ^= 1;
    }

    // Epilogue. C layout: VGPR r -> M = r + (lane<16 ? 0 : 8), N = lane&15.
    #pragma unroll
    for (int mi = 0; mi < 2; ++mi) {
        #pragma unroll
        for (int ni = 0; ni < 4; ++ni) {
            int col = n0 + wn * 64 + ni * 16 + mrow;
            #pragma unroll
            for (int r = 0; r < 8; ++r) {
                int row = m0 + wm * 32 + mi * 16 + r + ((lane < 16) ? 0 : 8);
                float v = acc[mi][ni][r];
                if constexpr (OUT_F32)
                    Cf[(size_t)row * N + col] = v + bias[col];
                else
                    Ch[(size_t)row * N + col] = (_Float16)v;
            }
        }
    }
}

// ---------------------------------------------------------------------------
// kv = k^T v per (b,h), fused with row L2-norm * gamma -> f16 KVn[bh][64][64].
// One block per bh; double-buffered LDS over the 128 K-iterations.
// A = k^T (M=d, K=n) via transposed LDS tile; B = v.
// ---------------------------------------------------------------------------
__global__ __launch_bounds__(256) void kv_norm_kernel(
    const _Float16* __restrict__ QKV, const float* __restrict__ gamma,
    _Float16* __restrict__ KVn, int seq) {
    __shared__ alignas(16) _Float16 Kt[2][64 * 32];   // [d][n]
    __shared__ alignas(16) _Float16 Vt[2][64 * 32];   // [e][n]
    __shared__ float kvbuf[64 * 64];
    __shared__ float ninv[64];

    const int bh = blockIdx.x;
    const int b = bh >> 4, h = bh & 15;
    const size_t rowBase = (size_t)b * seq;
    const int kcol = 1024 + h * 64, vcol = 2048 + h * 64;

    const int tid  = threadIdx.x;
    const int lane = tid & 31;
    const int wave = tid >> 5;
    const int mi   = wave >> 1;            // d-tile 0..3
    const int nb   = (wave & 1) * 2;       // e-tile base {0,2}
    const int mrow = lane & 15;
    const int kbA  = (lane < 16) ? 0 : 8;
    const int kbB  = (lane < 16) ? 0 : 8;

    const int n = tid >> 3, cg = tid & 7;   // staging geometry

    v8f acc[2] = {};

    // Prologue: stage tile nt=0 into buffer 0
    {
        const _Float16* rp = QKV + (rowBase + n) * 3072;
        v8h kk = *(const v8h*)(rp + kcol + cg * 8);
        v8h vv = *(const v8h*)(rp + vcol + cg * 8);
        #pragma unroll
        for (int j = 0; j < 8; ++j) {
            Kt[0][(cg * 8 + j) * 32 + n] = kk[j];
            Vt[0][(cg * 8 + j) * 32 + n] = vv[j];
        }
    }
    __syncthreads();

    int buf = 0;
    for (int nt = 0; nt < seq; nt += 32) {
        const bool nxt = (nt + 32) < seq;
        v8h kk, vv;
        if (nxt) {
            const _Float16* rp = QKV + (rowBase + nt + 32 + n) * 3072;
            kk = *(const v8h*)(rp + kcol + cg * 8);
            vv = *(const v8h*)(rp + vcol + cg * 8);
        }
        if (nt + 64 < seq)
            __builtin_prefetch(QKV + (rowBase + nt + 64 + n) * 3072 + kcol, 0, 0);

        // Compute on current buffer
        v16h a = ldfragA(&Kt[buf][(mi * 16 + mrow) * 32 + kbA]);
        #pragma unroll
        for (int t = 0; t < 2; ++t) {
            v16h bv = ldfragB(&Vt[buf][((nb + t) * 16 + mrow) * 32 + kbB]);
            acc[t] = wmma_f16(a, bv, acc[t]);
        }

        if (nxt) {
            #pragma unroll
            for (int j = 0; j < 8; ++j) {
                Kt[buf ^ 1][(cg * 8 + j) * 32 + n] = kk[j];
                Vt[buf ^ 1][(cg * 8 + j) * 32 + n] = vv[j];
            }
        }
        __syncthreads();
        buf ^= 1;
    }

    // Scatter accumulators into f32 LDS tile kv[d][e]
    #pragma unroll
    for (int t = 0; t < 2; ++t)
        #pragma unroll
        for (int r = 0; r < 8; ++r) {
            int row = mi * 16 + r + ((lane < 16) ? 0 : 8);
            int col = (nb + t) * 16 + mrow;
            kvbuf[row * 64 + col] = acc[t][r];
        }
    __syncthreads();

    if (tid < 64) {
        float s = 0.f;
        #pragma unroll 8
        for (int e = 0; e < 64; ++e) { float x = kvbuf[tid * 64 + e]; s += x * x; }
        ninv[tid] = gamma[h] * rsqrtf(s);
    }
    __syncthreads();

    #pragma unroll
    for (int i = 0; i < 16; ++i) {
        int idx = i * 256 + tid;
        KVn[(size_t)bh * 4096 + idx] = (_Float16)(kvbuf[idx] * ninv[idx >> 6]);
    }
}

// ---------------------------------------------------------------------------
// In-place q normalization: q[row, h*64 .. +63] *= gamma[h] / ||q_row_h||_2
// One thread per (row, head).
// ---------------------------------------------------------------------------
__global__ __launch_bounds__(256) void qnorm_kernel(
    _Float16* __restrict__ QKV, const float* __restrict__ gamma) {
    int t = blockIdx.x * 256 + threadIdx.x;
    int row = t >> 4, h = t & 15;
    _Float16* p = QKV + (size_t)row * 3072 + h * 64;
    v8h buf[8];
    float s = 0.f;
    #pragma unroll
    for (int i = 0; i < 8; ++i) {
        buf[i] = *(const v8h*)(p + i * 8);
        #pragma unroll
        for (int j = 0; j < 8; ++j) { float x = (float)buf[i][j]; s += x * x; }
    }
    float inv = gamma[h] * rsqrtf(s);
    #pragma unroll
    for (int i = 0; i < 8; ++i) {
        v8h o;
        #pragma unroll
        for (int j = 0; j < 8; ++j) o[j] = (_Float16)((float)buf[i][j] * inv);
        *(v8h*)(p + i * 8) = o;
    }
}

// ---------------------------------------------------------------------------
// out[:, h*64..] = q_n[:, h*64..] @ KVn[b,h]  (per-head 64x64 right factor).
// A fragments come straight from global (row-major q_n matches A layout).
// ---------------------------------------------------------------------------
__global__ __launch_bounds__(256) void attn_kernel(
    const _Float16* __restrict__ QKV, const _Float16* __restrict__ KVn,
    _Float16* __restrict__ Out) {
    __shared__ alignas(16) _Float16 Bs[64 * 64];   // transposed: [e][d]

    const int h  = blockIdx.y;
    const int m0 = blockIdx.x * 128;
    const int b  = m0 >> 12;                       // 4096 rows per batch
    const int bh = b * 16 + h;
    const int tid = threadIdx.x;

    #pragma unroll
    for (int i = 0; i < 2; ++i) {
        int lin = (tid * 2 + i) * 8;
        int d = lin >> 6, e0 = lin & 63;
        v8h v = *(const v8h*)(KVn + (size_t)bh * 4096 + d * 64 + e0);
        #pragma unroll
        for (int j = 0; j < 8; ++j) Bs[(e0 + j) * 64 + d] = v[j];
    }
    __syncthreads();

    const int lane = tid & 31;
    const int wave = tid >> 5;
    const int mrow = lane & 15;
    const int kbA  = (lane < 16) ? 0 : 8;
    const int kbB  = (lane < 16) ? 0 : 8;
    const int r0   = m0 + wave * 16 + mrow;

    v8f acc[4] = {};
    #pragma unroll
    for (int kt = 0; kt < 64; kt += 32) {
        v16h a = ldfragA(QKV + (size_t)r0 * 3072 + h * 64 + kt + kbA);
        #pragma unroll
        for (int ni = 0; ni < 4; ++ni) {
            v16h bv = ldfragB(Bs + (ni * 16 + mrow) * 64 + kt + kbB);
            acc[ni] = wmma_f16(a, bv, acc[ni]);
        }
    }

    #pragma unroll
    for (int ni = 0; ni < 4; ++ni) {
        int col = h * 64 + ni * 16 + mrow;
        #pragma unroll
        for (int r = 0; r < 8; ++r) {
            int row = m0 + wave * 16 + r + ((lane < 16) ? 0 : 8);
            Out[(size_t)row * 1024 + col] = (_Float16)acc[ni][r];
        }
    }
}

// ---------------------------------------------------------------------------
// Host-side launch
// ---------------------------------------------------------------------------
extern "C" void kernel_launch(void* const* d_in, const int* in_sizes, int n_in,
                              void* d_out, int out_size, void* d_ws, size_t ws_size,
                              hipStream_t stream) {
    (void)in_sizes; (void)n_in; (void)out_size; (void)ws_size;

    const float* x     = (const float*)d_in[0];   // [4,4096,1024]
    const float* Wqkv  = (const float*)d_in[1];   // [1024,3072]
    const float* Wo    = (const float*)d_in[2];   // [1024,1024]
    const float* bo    = (const float*)d_in[3];   // [1024]
    const float* gamma = (const float*)d_in[4];   // [16]
    float* out = (float*)d_out;                   // [4,4096,1024] f32

    const int M = 16384, DIM = 1024, N3 = 3072;

    char* ws = (char*)d_ws;
    _Float16* Xh    = (_Float16*)ws;                                   // 32 MB (reused as Attn)
    _Float16* Wqkvh = (_Float16*)(ws + (size_t)33554432);              //  6 MB
    _Float16* Woh   = (_Float16*)(ws + (size_t)33554432 + 6291456);    //  2 MB
    _Float16* QKVh  = (_Float16*)(ws + (size_t)33554432 + 6291456 + 2097152);             // 96 MB
    _Float16* KVnh  = (_Float16*)(ws + (size_t)33554432 + 6291456 + 2097152 + 100663296); // 0.5 MB
    _Float16* Attnh = Xh;  // x is dead once qkv is computed

    // 1) f32 -> f16 conversions
    cvt_f32_f16<<<(M * DIM) / 1024, 256, 0, stream>>>(x, Xh, M * DIM);
    cvt_f32_f16<<<(DIM * N3) / 1024, 256, 0, stream>>>(Wqkv, Wqkvh, DIM * N3);
    cvt_f32_f16<<<(DIM * DIM) / 1024, 256, 0, stream>>>(Wo, Woh, DIM * DIM);

    // 2) qkv = x @ W_qkv   (16384 x 3072 x 1024)
    gemm_f16_wmma<false><<<dim3(N3 / 128, M / 128), 256, 0, stream>>>(
        Xh, Wqkvh, QKVh, nullptr, nullptr, M, N3, DIM);

    // 3) kv = k^T v per head, fused norm -> KVn  (64 blocks)
    kv_norm_kernel<<<64, 256, 0, stream>>>(QKVh, gamma, KVnh, 4096);

    // 4) q normalization in place
    qnorm_kernel<<<(M * 16) / 256, 256, 0, stream>>>(QKVh, gamma);

    // 5) attn output per head: q_n @ KVn  -> Attnh (f16, [16384,1024])
    attn_kernel<<<dim3(M / 128, 16), 256, 0, stream>>>(QKVh, KVnh, Attnh);

    // 6) out = Attn @ W_o + b_o  (f32 output)
    gemm_f16_wmma<true><<<dim3(DIM / 128, M / 128), 256, 0, stream>>>(
        Attnh, Woh, nullptr, out, bo, M, DIM, DIM);
}
